// TrellisDenseMLP_21182778703897
// MI455X (gfx1250) — compile-verified, matrix operand order checked by reading, not run on hardware
//
#include <hip/hip_runtime.h>
#include <hip/hip_bf16.h>
#include <math.h>

#define HIDDEN 4096
#define INTER  11008
#define M_TOT  128

typedef __attribute__((ext_vector_type(16))) __bf16 v16bf;
typedef __attribute__((ext_vector_type(8)))  float  v8f;

union FragBF {
    v16bf v;
    unsigned int   u[8];
    unsigned short us[16];
    uint4 q[2];
};

// Pack two f32 -> packed bf16x2 (truncation) with a single v_perm_b32.
// Result[15:0] = hi16(f0), Result[31:16] = hi16(f1).
__device__ __forceinline__ unsigned int pack_bf16x2(float f0, float f1) {
    unsigned int u0 = __builtin_bit_cast(unsigned int, f0);
    unsigned int u1 = __builtin_bit_cast(unsigned int, f1);
    return __builtin_amdgcn_perm(u1, u0, 0x07060302u);
}

// f32 -> bf16 bits, round-to-nearest-even (epilogue only; cheap there)
__device__ __forceinline__ unsigned short cvt_bf16_rne(float f) {
    unsigned int u = __builtin_bit_cast(unsigned int, f);
    u += 0x7FFFu + ((u >> 16) & 1u);
    return (unsigned short)(u >> 16);
}

// Load a 16x32 bf16 A/B fragment chunk for this lane, converting from f32.
// p points at row_base + k0 + khalf; lane's elements are p[0..7] and p[16..23].
__device__ __forceinline__ v16bf load_frag_cvt_f32(const float* __restrict__ p) {
    float4 a0 = *(const float4*)(p);
    float4 a1 = *(const float4*)(p + 4);
    float4 b0 = *(const float4*)(p + 16);
    float4 b1 = *(const float4*)(p + 20);
    FragBF f;
    f.u[0] = pack_bf16x2(a0.x, a0.y);
    f.u[1] = pack_bf16x2(a0.z, a0.w);
    f.u[2] = pack_bf16x2(a1.x, a1.y);
    f.u[3] = pack_bf16x2(a1.z, a1.w);
    f.u[4] = pack_bf16x2(b0.x, b0.y);
    f.u[5] = pack_bf16x2(b0.z, b0.w);
    f.u[6] = pack_bf16x2(b1.x, b1.y);
    f.u[7] = pack_bf16x2(b1.z, b1.w);
    return f.v;
}

// Load a fragment chunk straight from packed bf16 (two 16B vector loads).
__device__ __forceinline__ v16bf load_frag_bf16(const unsigned short* __restrict__ p) {
    FragBF f;
    f.q[0] = *(const uint4*)(p);
    f.q[1] = *(const uint4*)(p + 16);
    return f.v;
}

__device__ __forceinline__ v8f wmma_bf16(v16bf a, v16bf b, v8f c) {
    return __builtin_amdgcn_wmma_f32_16x16x32_bf16(
        /*neg_a=*/false, a, /*neg_b=*/false, b,
        /*c_mod=*/(short)0, c, /*reuse_a=*/false, /*reuse_b=*/false);
}

#define PF_DIST 256  // floats ahead (~8 K-iterations, 1 KB per row)

// ---------------------------------------------------------------------------
// Stage 1: G[m, n] = silu(X Wg^T) * (X Wu^T), written as bf16 into d_ws.
// Grid: INTER/16 blocks x 128 threads (4 waves). Each wave: 2 M-tiles.
// ---------------------------------------------------------------------------
__global__ void __launch_bounds__(128)
swiglu_stage1(const float* __restrict__ x,
              const float* __restrict__ wg,
              const float* __restrict__ wu,
              unsigned short* __restrict__ gbuf) {
    const int lane     = threadIdx.x & 31;
    const int wave     = threadIdx.x >> 5;    // 0..3
    const int r        = lane & 15;           // fragment row (M or N) within tile
    const int khalf    = (lane >> 4) * 8;     // 0 or 8
    const int n0       = blockIdx.x * 16;     // INTER column strip

    v8f accg0 = {0,0,0,0,0,0,0,0};
    v8f accg1 = {0,0,0,0,0,0,0,0};
    v8f accu0 = {0,0,0,0,0,0,0,0};
    v8f accu1 = {0,0,0,0,0,0,0,0};

    const float* wg_row = wg + (size_t)(n0 + r) * HIDDEN + khalf;
    const float* wu_row = wu + (size_t)(n0 + r) * HIDDEN + khalf;
    const float* x_row0 = x + (size_t)((wave * 2 + 0) * 16 + r) * HIDDEN + khalf;
    const float* x_row1 = x + (size_t)((wave * 2 + 1) * 16 + r) * HIDDEN + khalf;

#pragma unroll 2
    for (int k0 = 0; k0 < HIDDEN; k0 += 32) {
        if (k0 + PF_DIST < HIDDEN) {
            __builtin_prefetch(wg_row + k0 + PF_DIST, 0, 1);
            __builtin_prefetch(wu_row + k0 + PF_DIST, 0, 1);
        }
        v16bf bg = load_frag_cvt_f32(wg_row + k0);
        v16bf bu = load_frag_cvt_f32(wu_row + k0);
        v16bf a0 = load_frag_cvt_f32(x_row0 + k0);
        v16bf a1 = load_frag_cvt_f32(x_row1 + k0);
        accg0 = wmma_bf16(a0, bg, accg0);
        accu0 = wmma_bf16(a0, bu, accu0);
        accg1 = wmma_bf16(a1, bg, accg1);
        accu1 = wmma_bf16(a1, bu, accu1);
    }

    // Epilogue: silu(gate)*up -> bf16 G. C layout: N = r, M = khalf + j.
    const int n = n0 + r;
    {
        const int mbase = (wave * 2 + 0) * 16 + khalf;
#pragma unroll
        for (int j = 0; j < 8; ++j) {
            float g = accg0[j], u = accu0[j];
            float s = g / (1.0f + __expf(-g));
            gbuf[(size_t)(mbase + j) * INTER + n] = cvt_bf16_rne(s * u);
        }
    }
    {
        const int mbase = (wave * 2 + 1) * 16 + khalf;
#pragma unroll
        for (int j = 0; j < 8; ++j) {
            float g = accg1[j], u = accu1[j];
            float s = g / (1.0f + __expf(-g));
            gbuf[(size_t)(mbase + j) * INTER + n] = cvt_bf16_rne(s * u);
        }
    }
}

// ---------------------------------------------------------------------------
// Stage 2: Out[m, h] = sum_i G[m, i] * Wd[h, i].  Grid: HIDDEN/16 x 128.
// ---------------------------------------------------------------------------
__global__ void __launch_bounds__(128)
swiglu_stage2(const unsigned short* __restrict__ gbuf,
              const float* __restrict__ wd,
              float* __restrict__ out) {
    const int lane  = threadIdx.x & 31;
    const int wave  = threadIdx.x >> 5;
    const int r     = lane & 15;
    const int khalf = (lane >> 4) * 8;
    const int n0    = blockIdx.x * 16;       // HIDDEN column strip

    v8f acc0 = {0,0,0,0,0,0,0,0};
    v8f acc1 = {0,0,0,0,0,0,0,0};

    const float*          wd_row = wd   + (size_t)(n0 + r) * INTER + khalf;
    const unsigned short* g_row0 = gbuf + (size_t)((wave * 2 + 0) * 16 + r) * INTER + khalf;
    const unsigned short* g_row1 = gbuf + (size_t)((wave * 2 + 1) * 16 + r) * INTER + khalf;

#pragma unroll 2
    for (int k0 = 0; k0 < INTER; k0 += 32) {
        if (k0 + PF_DIST < INTER) {
            __builtin_prefetch(wd_row + k0 + PF_DIST, 0, 1);
        }
        v16bf b  = load_frag_cvt_f32(wd_row + k0);
        v16bf a0 = load_frag_bf16(g_row0 + k0);
        v16bf a1 = load_frag_bf16(g_row1 + k0);
        acc0 = wmma_bf16(a0, b, acc0);
        acc1 = wmma_bf16(a1, b, acc1);
    }

    const int n = n0 + r;
    {
        const int mbase = (wave * 2 + 0) * 16 + khalf;
#pragma unroll
        for (int j = 0; j < 8; ++j)
            out[(size_t)(mbase + j) * HIDDEN + n] = acc0[j];
    }
    {
        const int mbase = (wave * 2 + 1) * 16 + khalf;
#pragma unroll
        for (int j = 0; j < 8; ++j)
            out[(size_t)(mbase + j) * HIDDEN + n] = acc1[j];
    }
}

extern "C" void kernel_launch(void* const* d_in, const int* in_sizes, int n_in,
                              void* d_out, int out_size, void* d_ws, size_t ws_size,
                              hipStream_t stream) {
    const float* x  = (const float*)d_in[0];
    const float* wg = (const float*)d_in[1];
    const float* wu = (const float*)d_in[2];
    const float* wd = (const float*)d_in[3];
    float* out = (float*)d_out;
    unsigned short* gbuf = (unsigned short*)d_ws;  // 128*11008 bf16 = 2.75 MB

    swiglu_stage1<<<INTER / 16, 128, 0, stream>>>(x, wg, wu, gbuf);
    swiglu_stage2<<<HIDDEN / 16, 128, 0, stream>>>(gbuf, wd, out);
}
